// NonLocalBlock_57062935494894
// MI455X (gfx1250) — matrix operand order repaired and checked by reading, not compiled, not run on hardware
//
#include <hip/hip_runtime.h>
#include <hip/hip_bf16.h>

typedef __bf16 bf16;
typedef __attribute__((ext_vector_type(16))) __bf16 v16bf;
typedef __attribute__((ext_vector_type(8)))  __bf16 v8bf;
typedef __attribute__((ext_vector_type(8)))  float  v8f;
typedef __attribute__((ext_vector_type(4)))  unsigned int u32x4;
typedef __attribute__((ext_vector_type(4)))  int  i32x4;
typedef __attribute__((ext_vector_type(8)))  int  i32x8;

// ---------------------------------------------------------------------------
// WMMA helper (CDNA5 wave32, v_wmma_f32_16x16x32_bf16)
// ---------------------------------------------------------------------------
static __device__ __forceinline__ v8f wmma_bf16(v16bf a, v16bf b, v8f c) {
  return __builtin_amdgcn_wmma_f32_16x16x32_bf16(false, a, false, b,
                                                 (short)0, c, false, false);
}

static __device__ __forceinline__ v8f vzero8() {
  v8f z;
#pragma unroll
  for (int i = 0; i < 8; ++i) z[i] = 0.0f;
  return z;
}

// A fragment 16x32 bf16 from row-major memory (ld in elements):
// lane L: row m=L&15, k = kb..kb+15, kb=((L>>4)<<4)
// -> two contiguous 16B chunks per lane (ds_load_b128 / global_load_b128).
static __device__ __forceinline__ v16bf loadA(const bf16* base, int ld, int lane) {
  const int m  = lane & 15;
  const int kb = (lane >> 4) << 4;
  v8bf lo = *(const v8bf*)(base + m * ld + kb);
  v8bf hi = *(const v8bf*)(base + m * ld + kb + 8);
  return __builtin_shufflevector(lo, hi, 0, 1, 2, 3, 4, 5, 6, 7,
                                 8, 9, 10, 11, 12, 13, 14, 15);
}

// B fragment 32x16 from n-major memory (element (k,n) at base[n*ld+k]):
// lane L: col n=L&15, k = kb..kb+15, kb=((L>>4)<<4) -> one contiguous 32B run.
static __device__ __forceinline__ v16bf loadB_nmajor(const bf16* base, int ld, int lane) {
  const int n  = lane & 15;
  const int kb = (lane >> 4) << 4;
  return *(const v16bf*)(base + n * ld + kb);
}

// B fragment 32x16 from k-major f32 memory (element (k,n) at base[k*ld+n]).
static __device__ __forceinline__ v16bf loadB_kmajor_f32(const float* base, int ld, int lane) {
  v16bf b;
  const int n  = lane & 15;
  const int kb = (lane >> 4) << 4;
#pragma unroll
  for (int j = 0; j < 16; ++j) b[j] = (bf16)base[(kb + j) * ld + n];
  return b;
}

// B fragment 32x16 from n-major f32 memory (element (k,n) at base[n*ld+k]).
static __device__ __forceinline__ v16bf loadB_nmajor_f32(const float* base, int ld, int lane) {
  v16bf b;
  const int n  = lane & 15;
  const int kb = (lane >> 4) << 4;
#pragma unroll
  for (int j = 0; j < 16; ++j) b[j] = (bf16)base[n * ld + kb + j];
  return b;
}

// A fragment from row-major f32 memory.
static __device__ __forceinline__ v16bf loadA_f32(const float* base, int ld, int lane) {
  v16bf a;
  const int m  = lane & 15;
  const int kb = (lane >> 4) << 4;
#pragma unroll
  for (int j = 0; j < 16; ++j) {
    const int k = kb + j;
    a[j] = (bf16)base[m * ld + k];
  }
  return a;
}

// ---------------------------------------------------------------------------
// Tensor Data Mover: 2D tile (bf16) global -> LDS.
// D# per CDNA5 ISA. Issued wave-uniform; tracked by
// TENSORcnt. Arity differs across toolchains -> dispatch on clang major.
// ---------------------------------------------------------------------------
static __device__ __forceinline__ void tdm_load_tile_bf16(
    unsigned lds_addr_bytes, const void* gaddr,
    unsigned tile_d0, unsigned tile_d1,
    unsigned tensor_d0, unsigned tensor_d1, unsigned stride_d0_elems) {
  const unsigned long long ga = (unsigned long long)(size_t)gaddr;
  u32x4 g0;
  g0[0] = 1u;                                   // count=1, user mode
  g0[1] = lds_addr_bytes;                       // LDS byte address
  g0[2] = (unsigned)(ga & 0xFFFFFFFFu);         // global_addr[31:0]
  g0[3] = (unsigned)((ga >> 32) & 0x01FFFFFFu)  // global_addr[56:32]
        | (2u << 30);                           // type=2 ("image")
  i32x8 g1;
  g1[0] = (int)(1u << 16);                      // data_size=1 (2B), wg_mask=0
  g1[1] = (int)((tensor_d0 & 0xFFFFu) << 16);   // tensor_dim0 lo16 @63:48
  g1[2] = (int)(((tensor_d0 >> 16) & 0xFFFFu) | ((tensor_d1 & 0xFFFFu) << 16));
  g1[3] = (int)(((tensor_d1 >> 16) & 0xFFFFu) | ((tile_d0 & 0xFFFFu) << 16));
  g1[4] = (int)(tile_d1 & 0xFFFFu);             // tile_dim1; tile_dim2=0
  g1[5] = (int)stride_d0_elems;                 // tensor_dim0_stride[31:0]
  g1[6] = 0;
  g1[7] = 0;
  i32x4 gz;
  gz[0] = 0; gz[1] = 0; gz[2] = 0; gz[3] = 0;
#if defined(__clang_major__) && (__clang_major__ >= 23)
  i32x8 gz8;
#pragma unroll
  for (int i = 0; i < 8; ++i) gz8[i] = 0;
  __builtin_amdgcn_tensor_load_to_lds(g0, g1, gz, gz, gz8, 0);
#else
  __builtin_amdgcn_tensor_load_to_lds(g0, g1, gz, gz, 0);
#endif
}

// ---------------------------------------------------------------------------
// Kernel 1: fused theta/phi/g 1x1 convs. All outputs n-major [B, hw, 128]
// bf16 (theta pre-scaled by 1/sqrt(128)). One wave: 16(M) x 64(N), K=256,
// 32 WMMAs.
// ---------------------------------------------------------------------------
__global__ __launch_bounds__(32) void k_qkv(
    const float* __restrict__ x,
    const float* __restrict__ w_theta, const float* __restrict__ b_theta,
    const float* __restrict__ w_phi,   const float* __restrict__ b_phi,
    const float* __restrict__ w_g,     const float* __restrict__ b_g,
    bf16* __restrict__ thetaT, bf16* __restrict__ phiT, bf16* __restrict__ gT)
{
  const int HW = 4096, C = 256, C2 = 128;
  const int lane = threadIdx.x & 31;
  const int n0 = blockIdx.x * 64;
  const int ty = blockIdx.y;       // 0..23
  const int mat = ty >> 3;         // 0=theta 1=phi 2=g
  const int m0 = (ty & 7) << 4;    // c2 tile base
  const int b = blockIdx.z;

  const float* W   = (mat == 0) ? w_theta : (mat == 1) ? w_phi : w_g;
  const float* Bv  = (mat == 0) ? b_theta : (mat == 1) ? b_phi : b_g;
  bf16*        dst = (mat == 0) ? thetaT  : (mat == 1) ? phiT  : gT;
  const float  scl = (mat == 0) ? 0.0883883476483184405f : 1.0f; // 1/sqrt(128)
  const float* xb = x + (size_t)b * C * HW;

  v8f acc[4];
#pragma unroll
  for (int nt = 0; nt < 4; ++nt) acc[nt] = vzero8();

  for (int k0 = 0; k0 < C; k0 += 32) {
    v16bf a = loadA_f32(W + m0 * C + k0, C, lane);
#pragma unroll
    for (int nt = 0; nt < 4; ++nt) {
      v16bf bb = loadB_kmajor_f32(xb + (size_t)k0 * HW + n0 + nt * 16, HW, lane);
      acc[nt] = wmma_bf16(a, bb, acc[nt]);
    }
  }

  const size_t bo = (size_t)b * C2 * HW;
#pragma unroll
  for (int nt = 0; nt < 4; ++nt) {
#pragma unroll
    for (int r = 0; r < 8; ++r) {
      const int c2 = m0 + r + ((lane >> 4) << 3);
      const int n  = n0 + nt * 16 + (lane & 15);
      dst[bo + (size_t)n * C2 + c2] = (bf16)((acc[nt][r] + Bv[c2]) * scl);
    }
  }
}

// ---------------------------------------------------------------------------
// Kernel 2: flash attention, 128 threads (4 waves), 64 query rows per block.
// K tile staged by the Tensor Data Mover (tensor_load_to_lds + TENSORcnt);
// V tile staged transposed so PV B-fragments are contiguous 32B LDS runs.
// Per 32-key tile per wave: 8 WMMA (S=QK) + online softmax + 8 WMMA (O+=PV).
// ---------------------------------------------------------------------------
__global__ __launch_bounds__(128) void k_attn(
    const bf16* __restrict__ thetaT, const bf16* __restrict__ phiT,
    const bf16* __restrict__ gT, bf16* __restrict__ yT)
{
  const int HW = 4096, C2 = 128;
  __shared__ __align__(32) bf16 Qs[64 * 128];    // 16 KB, [n][c2]
  __shared__ __align__(32) bf16 Ks[32 * 128];    //  8 KB, [m][c2] (TDM target)
  __shared__ __align__(32) bf16 VsT[128 * 32];   //  8 KB, [c2][m]
  __shared__ __align__(32) bf16 Ps[4 * 16 * 32]; //  4 KB, per-wave P staging

  const int tid = threadIdx.x, lane = tid & 31, wv = tid >> 5;
  const int b = blockIdx.y, n0 = blockIdx.x * 64;
  const size_t bo = (size_t)b * C2 * HW;
  const bf16* phiTb = phiT + bo;
  const bf16* gTb   = gT + bo;

  for (int i = tid; i < 64 * 128; i += 128)
    Qs[i] = thetaT[bo + (size_t)(n0 + (i >> 7)) * C2 + (i & 127)];
  __syncthreads();

  v8f o[8];
#pragma unroll
  for (int nt = 0; nt < 8; ++nt) o[nt] = vzero8();
  float mrow[8], lrow[8];
#pragma unroll
  for (int r = 0; r < 8; ++r) { mrow[r] = -1e30f; lrow[r] = 0.0f; }

  const bf16* Qw = Qs + wv * 16 * 128;
  bf16* Pw = Ps + wv * 512;
  const unsigned ksLds = (unsigned)(unsigned long long)(const void*)Ks;

  for (int kt = 0; kt < HW / 32; ++kt) {
    const int m0 = kt * 32;

    // --- K tile [32 m][128 c2] via Tensor Data Mover (wave 0 issues) ---
    if (wv == 0) {
      tdm_load_tile_bf16(ksLds, phiTb + (size_t)m0 * C2,
                         /*tile*/ 128, 32, /*tensor*/ 128, HW, /*stride*/ 128);
      __builtin_amdgcn_s_wait_tensorcnt(0);
    }
    // --- V tile transposed: VsT[c2][m] = gT[m0+m][c2] ---
#pragma unroll
    for (int it = 0; it < 4; ++it) {
      const int idx = (it * 128 + tid) * 8;   // 8 bf16 per thread per step
      const int m = idx >> 7, c = idx & 127;
      v8bf v = *(const v8bf*)(gTb + (size_t)(m0 + m) * C2 + c);
#pragma unroll
      for (int j = 0; j < 8; ++j) VsT[(c + j) * 32 + m] = v[j];
    }
    // prefetch next tiles into cache hierarchy (global_prefetch_b8)
    if (kt + 1 < HW / 32) {
      __builtin_prefetch(phiTb + (size_t)(m0 + 32) * C2, 0, 3);
      __builtin_prefetch(gTb + (size_t)(m0 + 32) * C2, 0, 3);
    }
    __syncthreads();

    // --- S[16x32] = Q[16x128] * K[128x32]; scale folded into theta ---
    v8f s0 = vzero8(), s1 = vzero8();
#pragma unroll
    for (int kk = 0; kk < 4; ++kk) {
      v16bf a   = loadA(Qw + kk * 32, 128, lane);
      v16bf bb0 = loadB_nmajor(Ks + kk * 32, 128, lane);            // m=0..15
      v16bf bb1 = loadB_nmajor(Ks + 16 * 128 + kk * 32, 128, lane); // m=16..31
      s0 = wmma_bf16(a, bb0, s0);
      s1 = wmma_bf16(a, bb1, s1);
    }

    // --- online softmax (C/D rows live across 16-lane halves) ---
#pragma unroll
    for (int r = 0; r < 8; ++r) {
      float t = fmaxf(s0[r], s1[r]);
#pragma unroll
      for (int mask = 1; mask < 16; mask <<= 1)
        t = fmaxf(t, __shfl_xor(t, mask, 32));
      const float nm = fmaxf(mrow[r], t);
      const float al = __expf(mrow[r] - nm);
      const float p0 = __expf(s0[r] - nm);
      const float p1 = __expf(s1[r] - nm);
      float rs = p0 + p1;
#pragma unroll
      for (int mask = 1; mask < 16; mask <<= 1)
        rs += __shfl_xor(rs, mask, 32);
      lrow[r] = lrow[r] * al + rs;
      mrow[r] = nm;
#pragma unroll
      for (int nt = 0; nt < 8; ++nt) o[nt][r] = o[nt][r] * al;
      const int row = r + ((lane >> 4) << 3);
      Pw[row * 32 + (lane & 15)]      = (bf16)p0;  // C/D layout -> LDS
      Pw[row * 32 + 16 + (lane & 15)] = (bf16)p1;
    }
    asm volatile("s_wait_dscnt 0" ::: "memory");   // P store -> P load order

    // --- O[16x128] += P[16x32] * V[32x128] ---
    v16bf ap = loadA(Pw, 32, lane);
#pragma unroll
    for (int nt = 0; nt < 8; ++nt) {
      v16bf bv = loadB_nmajor(VsT + (nt * 16) * 32, 32, lane);
      o[nt] = wmma_bf16(ap, bv, o[nt]);
    }
    __syncthreads();
  }

  // normalize and store y^T [B, hw, C2] bf16
#pragma unroll
  for (int r = 0; r < 8; ++r) {
    const float inv = 1.0f / lrow[r];
    const int n = n0 + wv * 16 + r + ((lane >> 4) << 3);
#pragma unroll
    for (int nt = 0; nt < 8; ++nt) {
      const int c2 = nt * 16 + (lane & 15);
      yT[bo + (size_t)n * C2 + c2] = (bf16)(o[nt][r] * inv);
    }
  }
}

// ---------------------------------------------------------------------------
// Kernel 3: out-conv (K=128) + BatchNorm(eval) + residual, fused epilogue.
// One wave: 16(n) x 64(c), 16 WMMAs.
// ---------------------------------------------------------------------------
__global__ __launch_bounds__(32) void k_out(
    const bf16* __restrict__ yT, const float* __restrict__ w_out,
    const float* __restrict__ b_out, const float* __restrict__ gamma,
    const float* __restrict__ beta, const float* __restrict__ rmean,
    const float* __restrict__ rvar, const float* __restrict__ x,
    float* __restrict__ out)
{
  const int HW = 4096, C = 256, C2 = 128;
  const int lane = threadIdx.x & 31;
  const int n0 = blockIdx.x * 16, c0 = blockIdx.y * 64, b = blockIdx.z;
  const bf16* yb = yT + (size_t)b * C2 * HW;

  v8f acc[4];
#pragma unroll
  for (int ct = 0; ct < 4; ++ct) acc[ct] = vzero8();

  for (int k0 = 0; k0 < C2; k0 += 32) {
    v16bf a = loadA(yb + (size_t)n0 * C2 + k0, C2, lane);
#pragma unroll
    for (int ct = 0; ct < 4; ++ct) {
      // B(k=c2, n=c) = w_out[c, c2] : n-major with ld=C2
      v16bf bb = loadB_nmajor_f32(w_out + (size_t)(c0 + ct * 16) * C2 + k0, C2, lane);
      acc[ct] = wmma_bf16(a, bb, acc[ct]);
    }
  }

#pragma unroll
  for (int ct = 0; ct < 4; ++ct) {
    const int c = c0 + ct * 16 + (lane & 15);
    const float inv = gamma[c] * rsqrtf(rvar[c] + 1e-5f);
    const float sh  = beta[c] - rmean[c] * inv;
    const float bc  = b_out[c];
#pragma unroll
    for (int r = 0; r < 8; ++r) {
      const int n = n0 + r + ((lane >> 4) << 3);
      const size_t idx = (size_t)b * C * HW + (size_t)c * HW + n;
      out[idx] = x[idx] + (acc[ct][r] + bc) * inv + sh;
    }
  }
}

// ---------------------------------------------------------------------------
extern "C" void kernel_launch(void* const* d_in, const int* in_sizes, int n_in,
                              void* d_out, int out_size, void* d_ws, size_t ws_size,
                              hipStream_t stream) {
  const float* x       = (const float*)d_in[0];
  const float* w_theta = (const float*)d_in[1];
  const float* b_theta = (const float*)d_in[2];
  const float* w_phi   = (const float*)d_in[3];
  const float* b_phi   = (const float*)d_in[4];
  const float* w_g     = (const float*)d_in[5];
  const float* b_g     = (const float*)d_in[6];
  const float* w_out   = (const float*)d_in[7];
  const float* b_out   = (const float*)d_in[8];
  const float* gamma   = (const float*)d_in[9];
  const float* beta    = (const float*)d_in[10];
  const float* rmean   = (const float*)d_in[11];
  const float* rvar    = (const float*)d_in[12];
  float* out = (float*)d_out;

  const size_t SZ = (size_t)8 * 4096 * 128;  // elems per [B, hw, C2] tensor
  bf16* thetaT = (bf16*)d_ws;                // 8 MB
  bf16* phiT   = thetaT + SZ;                // 8 MB
  bf16* gT     = phiT + SZ;                  // 8 MB
  bf16* yT     = gT + SZ;                    // 8 MB   (total 32 MB)

  dim3 g1(64, 24, 8);   // hw/64 x (3 mats * 128/16) x B
  k_qkv<<<g1, dim3(32), 0, stream>>>(x, w_theta, b_theta, w_phi, b_phi,
                                     w_g, b_g, thetaT, phiT, gT);

  dim3 g2(64, 8);       // hw/64 x B
  k_attn<<<g2, dim3(128), 0, stream>>>(thetaT, phiT, gT, yT);

  dim3 g3(256, 4, 8);   // hw/16 x C/64 x B
  k_out<<<g3, dim3(32), 0, stream>>>(yT, w_out, b_out, gamma, beta,
                                     rmean, rvar, x, out);
}
